// MYTransformer_66846870995393
// MI455X (gfx1250) — compile-verified, matrix operand order checked by reading, not run on hardware
//
#include <hip/hip_runtime.h>

// ---------------------------------------------------------------------------
// Types
// ---------------------------------------------------------------------------
typedef __attribute__((ext_vector_type(16))) __bf16        bf16x16;
typedef __attribute__((ext_vector_type(8)))  float         f32x8;
typedef __attribute__((ext_vector_type(4)))  float         f32x4;
typedef __attribute__((ext_vector_type(4)))  unsigned int  u32x4;
typedef unsigned short u16;
typedef unsigned int   u32;

#define DM   1024          // model dim
#define HH   16            // heads
#define DH   64            // head dim
#define FFD  4096          // ffn hidden
#define NLAY 6
#define BB   4
#define SS   1024
#define M4   (BB*SS)       // 4096 flattened tokens
#define D3   (3*DM)
#define ZBH  (BB*HH)       // 64

#define LRS  40            // LDS row stride in u16 (80B: bank-conflict-free frags)

// ---------------------------------------------------------------------------
// Helpers
// ---------------------------------------------------------------------------
__device__ __forceinline__ u16 f2bf(float f) {
    u32 u = __float_as_uint(f);
    u32 r = u + 0x7FFFu + ((u >> 16) & 1u);     // round-to-nearest-even
    return (u16)(r >> 16);
}
__device__ __forceinline__ u32 pack2(float a, float b) {
    return (u32)f2bf(a) | ((u32)f2bf(b) << 16);
}

// Two async global->LDS 16B copies: covers 32 contiguous bytes.
// INST_OFFSET is added to BOTH the LDS and global address (ISA 08 §4.4),
// so one (lds,goff) pair + offset:16 covers the second chunk.
__device__ __forceinline__ void async_ld_b128x2(unsigned lds, unsigned goff,
                                                const void* base) {
    asm volatile("global_load_async_to_lds_b128 %0, %1, %2\n\t"
                 "global_load_async_to_lds_b128 %0, %1, %2 offset:16"
                 :: "v"(lds), "v"(goff), "s"(base)
                 : "memory");
}
__device__ __forceinline__ void wait_async0() {
    asm volatile("s_wait_asynccnt 0x0" ::: "memory");
}
__device__ __forceinline__ unsigned lds_off(const void* p) {
    return (unsigned)(unsigned long long)p;   // low 32 bits = LDS offset
}

// Load one 16x32 bf16 fragment (A or B operand) from LDS, CDNA5 16-bit layout:
// lane = {r=lane&15 row, half=lane>>4}:
//   VGPR p<4 : K = 2p       + 8*half   (8 contiguous bf16 -> one b128)
//   VGPR p>=4: K = 16+2(p-4)+ 8*half   (8 contiguous bf16 -> one b128)
__device__ __forceinline__ bf16x16 ld_frag(const u16* lds, int row, int half) {
    union { bf16x16 v; u32x4 q[2]; } f;
    const u16* p = lds + row * LRS + half * 8;
    f.q[0] = *(const u32x4*)(p);
    f.q[1] = *(const u32x4*)(p + 16);
    return f.v;
}

// ---------------------------------------------------------------------------
// GEMM parameter block
// ---------------------------------------------------------------------------
struct GemmP {
    const u16*   A;  long lda, sAb, sAh;   // bf16 activations
    const void*  B;  long ldb, sBb, sBh;   // fp32 weights or bf16
    const float* bias;
    const float* resid; long ldr;
    float* Cf; u16* Cb;  long ldc, sCb, sCh;
    int M, N, K, Hz, relu;
    float scale;
};

// ---------------------------------------------------------------------------
// C = A * B^T  (A: [M,K] bf16 k-contig, B: [N,K] k-contig fp32 or bf16)
// Block tile 128x128x32, 8 waves: wave grid 4(M) x 2(N), wave tile 32x64.
// Double-buffered LDS; bf16 operands staged with async-to-LDS.
// ---------------------------------------------------------------------------
template<bool BBF>
__global__ __launch_bounds__(256) void gemm_nt_k(GemmP p) {
    __shared__ u16 lA[2][128 * LRS];
    __shared__ u16 lB[2][128 * LRS];

    const int tid  = threadIdx.x;
    const int wid  = tid >> 5, lane = tid & 31;
    const int half = lane >> 4, l16 = lane & 15;
    const int zb = blockIdx.z / p.Hz, zh = blockIdx.z % p.Hz;
    const int m0 = blockIdx.y * 128, n0 = blockIdx.x * 128;
    const int wm = wid >> 1, wn = wid & 1;

    const u16*   Ag   = p.A + (long)zb * p.sAb + (long)zh * p.sAh;
    const u16*   Bg16 = (const u16*)p.B + (long)zb * p.sBb + (long)zh * p.sBh;
    const float* Bgf  = (const float*)p.B + (long)zb * p.sBb + (long)zh * p.sBh;

    f32x8 acc[2][4];
#pragma unroll
    for (int i = 0; i < 2; ++i)
#pragma unroll
        for (int j = 0; j < 4; ++j) acc[i][j] = f32x8{0,0,0,0,0,0,0,0};

    const int lrow = tid >> 1;            // 0..127
    const int lk   = (tid & 1) * 16;      // 0 or 16

    auto issueA = [&](int k0, int buf) {
        async_ld_b128x2(lds_off(&lA[buf][lrow * LRS + lk]),
                        (unsigned)(((long)(m0 + lrow) * p.lda + k0 + lk) * 2), Ag);
    };
    auto issueB16 = [&](int k0, int buf) {
        async_ld_b128x2(lds_off(&lB[buf][lrow * LRS + lk]),
                        (unsigned)(((long)(n0 + lrow) * p.ldb + k0 + lk) * 2), Bg16);
    };
    f32x4 rb0, rb1, rb2, rb3;
    auto loadBf32 = [&](int k0) {
        const float* bp = Bgf + (long)(n0 + lrow) * p.ldb + k0 + lk;
        rb0 = *(const f32x4*)(bp);
        rb1 = *(const f32x4*)(bp + 4);
        rb2 = *(const f32x4*)(bp + 8);
        rb3 = *(const f32x4*)(bp + 12);
        __builtin_prefetch(bp + 32, 0, 0);           // global_prefetch_b8
    };
    auto storeBf32 = [&](int buf) {
        u32x4 q0 = { pack2(rb0[0],rb0[1]), pack2(rb0[2],rb0[3]),
                     pack2(rb1[0],rb1[1]), pack2(rb1[2],rb1[3]) };
        u32x4 q1 = { pack2(rb2[0],rb2[1]), pack2(rb2[2],rb2[3]),
                     pack2(rb3[0],rb3[1]), pack2(rb3[2],rb3[3]) };
        *(u32x4*)&lB[buf][lrow * LRS + lk]     = q0;
        *(u32x4*)&lB[buf][lrow * LRS + lk + 8] = q1;
    };

    const int nIt = p.K >> 5;
    // ---- prologue: stage tile 0 into buffer 0 ----
    issueA(0, 0);
    if (BBF) issueB16(0, 0);
    else     { loadBf32(0); storeBf32(0); }
    wait_async0();
    __syncthreads();

    for (int it = 0; it < nIt; ++it) {
        const int cur = it & 1, nb = cur ^ 1;
        const bool hasNext = (it + 1) < nIt;
        if (hasNext) {
            issueA((it + 1) * 32, nb);
            if (BBF) issueB16((it + 1) * 32, nb);
            else     loadBf32((it + 1) * 32);
        }
        // ---- compute on current buffer while next tile is in flight ----
        bf16x16 af[2], bfg[4];
#pragma unroll
        for (int i = 0; i < 2; ++i) af[i]  = ld_frag(&lA[cur][0], wm * 32 + i * 16 + l16, half);
#pragma unroll
        for (int j = 0; j < 4; ++j) bfg[j] = ld_frag(&lB[cur][0], wn * 64 + j * 16 + l16, half);
#pragma unroll
        for (int i = 0; i < 2; ++i)
#pragma unroll
            for (int j = 0; j < 4; ++j)
                acc[i][j] = __builtin_amdgcn_wmma_f32_16x16x32_bf16(
                    false, af[i], false, bfg[j], (short)0, acc[i][j], false, false);
        if (hasNext && !BBF) storeBf32(nb);
        wait_async0();                 // my async writes to nb are done
        __syncthreads();               // (+dscnt) all waves' staging visible
    }

    // ---- epilogue ----
    const long cz = (long)zb * p.sCb + (long)zh * p.sCh;
#pragma unroll
    for (int i = 0; i < 2; ++i)
#pragma unroll
        for (int j = 0; j < 4; ++j) {
            const int n = n0 + wn * 64 + j * 16 + l16;
            float bv = p.bias ? p.bias[n] : 0.0f;
#pragma unroll
            for (int r = 0; r < 8; ++r) {
                const int m = m0 + wm * 32 + i * 16 + r + 8 * half;
                float v = acc[i][j][r] * p.scale + bv;
                if (p.resid) v += p.resid[(long)m * p.ldr + n];
                if (p.relu)  v = fmaxf(v, 0.0f);
                const long co = cz + (long)m * p.ldc + n;
                if (p.Cf) p.Cf[co] = v;
                if (p.Cb) p.Cb[co] = f2bf(v);
            }
        }
}

// ---------------------------------------------------------------------------
// C = A * B  (A: [M,K] bf16, B: [K,N] fp32 n-contig)  — used for P @ V.
// Block tile 128x64x32, 8 waves: 4(M) x 2(N), wave tile 32x32.
// ---------------------------------------------------------------------------
__global__ __launch_bounds__(256) void gemm_nn_k(GemmP p) {
    __shared__ u16 lA[2][128 * LRS];
    __shared__ u16 lB[2][64 * LRS];

    const int tid  = threadIdx.x;
    const int wid  = tid >> 5, lane = tid & 31;
    const int half = lane >> 4, l16 = lane & 15;
    const int zb = blockIdx.z / p.Hz, zh = blockIdx.z % p.Hz;
    const int m0 = blockIdx.y * 128, n0 = blockIdx.x * 64;
    const int wm = wid >> 1, wn = wid & 1;

    const u16*   Ag = p.A + (long)zb * p.sAb + (long)zh * p.sAh;
    const float* Bg = (const float*)p.B + (long)zb * p.sBb + (long)zh * p.sBh;

    f32x8 acc[2][2];
#pragma unroll
    for (int i = 0; i < 2; ++i)
#pragma unroll
        for (int j = 0; j < 2; ++j) acc[i][j] = f32x8{0,0,0,0,0,0,0,0};

    const int lrow = tid >> 1;
    const int lk   = (tid & 1) * 16;

    auto issueA = [&](int k0, int buf) {
        async_ld_b128x2(lds_off(&lA[buf][lrow * LRS + lk]),
                        (unsigned)(((long)(m0 + lrow) * p.lda + k0 + lk) * 2), Ag);
    };
    f32x4 rb0, rb1;
    auto loadBf32 = [&](int k0) {
        const int c0 = tid, c1 = tid + 256;      // f32x4 chunk ids 0..511
        rb0 = *(const f32x4*)(Bg + (long)(k0 + (c0 >> 4)) * p.ldb + n0 + (c0 & 15) * 4);
        rb1 = *(const f32x4*)(Bg + (long)(k0 + (c1 >> 4)) * p.ldb + n0 + (c1 & 15) * 4);
    };
    auto storeBf32 = [&](int buf) {
        const int c0 = tid, c1 = tid + 256;
        const int k0l = c0 >> 4, nc0 = (c0 & 15) * 4;
        const int k1l = c1 >> 4, nc1 = (c1 & 15) * 4;
#pragma unroll
        for (int q = 0; q < 4; ++q) {
            lB[buf][(nc0 + q) * LRS + k0l] = f2bf(rb0[q]);
            lB[buf][(nc1 + q) * LRS + k1l] = f2bf(rb1[q]);
        }
    };

    const int nIt = p.K >> 5;
    issueA(0, 0);
    loadBf32(0); storeBf32(0);
    wait_async0();
    __syncthreads();

    for (int it = 0; it < nIt; ++it) {
        const int cur = it & 1, nb = cur ^ 1;
        const bool hasNext = (it + 1) < nIt;
        if (hasNext) { issueA((it + 1) * 32, nb); loadBf32((it + 1) * 32); }

        bf16x16 af[2], bfg[2];
#pragma unroll
        for (int i = 0; i < 2; ++i) af[i]  = ld_frag(&lA[cur][0], wm * 32 + i * 16 + l16, half);
#pragma unroll
        for (int j = 0; j < 2; ++j) bfg[j] = ld_frag(&lB[cur][0], wn * 32 + j * 16 + l16, half);
#pragma unroll
        for (int i = 0; i < 2; ++i)
#pragma unroll
            for (int j = 0; j < 2; ++j)
                acc[i][j] = __builtin_amdgcn_wmma_f32_16x16x32_bf16(
                    false, af[i], false, bfg[j], (short)0, acc[i][j], false, false);

        if (hasNext) storeBf32(nb);
        wait_async0();
        __syncthreads();
    }

    const long cz = (long)zb * p.sCb + (long)zh * p.sCh;
#pragma unroll
    for (int i = 0; i < 2; ++i)
#pragma unroll
        for (int j = 0; j < 2; ++j) {
            const int n = n0 + wn * 32 + j * 16 + l16;
            float bv = p.bias ? p.bias[n] : 0.0f;
#pragma unroll
            for (int r = 0; r < 8; ++r) {
                const int m = m0 + wm * 32 + i * 16 + r + 8 * half;
                float v = acc[i][j][r] * p.scale + bv;
                if (p.resid) v += p.resid[(long)m * p.ldr + n];
                if (p.relu)  v = fmaxf(v, 0.0f);
                const long co = cz + (long)m * p.ldc + n;
                if (p.Cf) p.Cf[co] = v;
                if (p.Cb) p.Cb[co] = f2bf(v);
            }
        }
}

// ---------------------------------------------------------------------------
// LayerNorm (torch variant: unbiased std ddof=1, eps added to std).
// One 256-thread block per row of 1024. Writes f32 + bf16 mirror.
// ---------------------------------------------------------------------------
__global__ __launch_bounds__(256) void ln_k(const float* X, const float* ga,
                                            const float* gb, float* Of, u16* Ob) {
    const long row = blockIdx.x;
    const float* x = X + row * DM;
    const int t = threadIdx.x;
    float v[4]; float s = 0.f, sq = 0.f;
#pragma unroll
    for (int i = 0; i < 4; ++i) {
        v[i] = x[t + 256 * i]; s += v[i]; sq += v[i] * v[i];
    }
    __shared__ float r1[256], r2[256];
    r1[t] = s; r2[t] = sq; __syncthreads();
    for (int st = 128; st > 0; st >>= 1) {
        if (t < st) { r1[t] += r1[t + st]; r2[t] += r2[t + st]; }
        __syncthreads();
    }
    const float mu  = r1[0] * (1.0f / DM);
    float var = (r2[0] - (float)DM * mu * mu) * (1.0f / (DM - 1));
    var = fmaxf(var, 0.0f);
    const float inv = 1.0f / (sqrtf(var) + 1e-6f);
#pragma unroll
    for (int i = 0; i < 4; ++i) {
        const int c = t + 256 * i;
        const float o = ga[c] * (v[i] - mu) * inv + gb[c];
        Of[row * DM + c] = o;
        Ob[row * DM + c] = f2bf(o);
    }
}

// ---------------------------------------------------------------------------
// Softmax over a row of 1024 scores (optional causal mask), writes bf16 P.
// grid = (S, B*H)
// ---------------------------------------------------------------------------
__global__ __launch_bounds__(256) void softmax_k(const float* Sc, u16* P, int causal) {
    const int q = blockIdx.x;
    const long z = blockIdx.y;
    const float* s = Sc + (z * SS + q) * (long)SS;
    u16* p = P + (z * SS + q) * (long)SS;
    const int limit = causal ? (q + 1) : SS;
    const int t = threadIdx.x;

    float v[4]; float mx = -3.4e38f;
#pragma unroll
    for (int i = 0; i < 4; ++i) {
        const int c = t + 256 * i;
        v[i] = (c < limit) ? s[c] : -3.4e38f;
        mx = fmaxf(mx, v[i]);
    }
    __shared__ float red[256];
    red[t] = mx; __syncthreads();
    for (int st = 128; st > 0; st >>= 1) {
        if (t < st) red[t] = fmaxf(red[t], red[t + st]);
        __syncthreads();
    }
    const float M = red[0];
    __syncthreads();
    float e[4]; float sum = 0.f;
#pragma unroll
    for (int i = 0; i < 4; ++i) {
        const int c = t + 256 * i;
        e[i] = (c < limit) ? __expf(v[i] - M) : 0.0f;
        sum += e[i];
    }
    red[t] = sum; __syncthreads();
    for (int st = 128; st > 0; st >>= 1) {
        if (t < st) red[t] += red[t + st];
        __syncthreads();
    }
    const float inv = 1.0f / red[0];
#pragma unroll
    for (int i = 0; i < 4; ++i) {
        const int c = t + 256 * i;
        p[c] = f2bf(e[i] * inv);
    }
}

// ---------------------------------------------------------------------------
// Embedding gather * sqrt(D) + sinusoidal positional encoding (computed).
// ---------------------------------------------------------------------------
__global__ __launch_bounds__(256) void embed_k(const int* tok, const float* emb, float* X) {
    const long row = blockIdx.x;
    const int  s   = (int)(row & (SS - 1));
    const int  tk  = tok[row];
    const int  t   = threadIdx.x;
#pragma unroll
    for (int i = 0; i < 4; ++i) {
        const int c  = t + 256 * i;
        const int pr = c & ~1;
        const float div = __expf((float)pr * (-9.210340371976184f / (float)DM));
        const float arg = (float)s * div;
        const float pe  = (c & 1) ? __cosf(arg) : __sinf(arg);
        X[row * DM + c] = emb[(long)tk * DM + c] * 32.0f + pe;
    }
}

// ---------------------------------------------------------------------------
// Host orchestration
// ---------------------------------------------------------------------------
static GemmP GP(const u16* A, long lda, long sAb, long sAh,
                const void* B, long ldb, long sBb, long sBh,
                const float* bias, const float* resid, long ldr,
                float* Cf, u16* Cb, long ldc, long sCb, long sCh,
                int M, int N, int K, int Hz, int relu, float scale) {
    GemmP p{A, lda, sAb, sAh, B, ldb, sBb, sBh, bias, resid, ldr,
            Cf, Cb, ldc, sCb, sCh, M, N, K, Hz, relu, scale};
    return p;
}

static void run_gemm_nt(hipStream_t s, bool bbf, const GemmP& p, int Z) {
    dim3 g(p.N / 128, p.M / 128, Z), b(256, 1, 1);
    if (bbf) gemm_nt_k<true><<<g, b, 0, s>>>(p);
    else     gemm_nt_k<false><<<g, b, 0, s>>>(p);
}
static void run_gemm_nn(hipStream_t s, const GemmP& p, int Z) {
    dim3 g(p.N / 64, p.M / 128, Z), b(256, 1, 1);
    gemm_nn_k<<<g, b, 0, s>>>(p);
}

// input indices (setup_inputs dict order, nested dicts flattened in order)
enum {
    I_SRC = 0, I_TGT, I_EMB_S, I_EMB_T,
    E_QKV_W, E_QKV_B, E_PROJ_W, E_PROJ_B, E_W1, E_B1, E_W2, E_B2,
    E_LN0A, E_LN0B, E_LN1A, E_LN1B, E_LN2A, E_LN2B,
    D_QKV_W, D_QKV_B, D_SPROJ_W, D_SPROJ_B, D_WQ_W, D_WQ_B, D_WK_W, D_WK_B,
    D_WV_W, D_WV_B, D_CPROJ_W, D_CPROJ_B, D_W1, D_B1, D_W2, D_B2,
    D_LN0A, D_LN0B, D_LN1A, D_LN1B, D_LN2A, D_LN2B, D_LN3A, D_LN3B
};

extern "C" void kernel_launch(void* const* d_in, const int* in_sizes, int n_in,
                              void* d_out, int out_size, void* d_ws, size_t ws_size,
                              hipStream_t stream) {
    (void)in_sizes; (void)n_in; (void)out_size; (void)ws_size;

    const int*   src   = (const int*)d_in[I_SRC];
    const int*   tgt   = (const int*)d_in[I_TGT];
    const float* embS  = (const float*)d_in[I_EMB_S];
    const float* embT  = (const float*)d_in[I_EMB_T];
    auto F = [&](int i) { return (const float*)d_in[i]; };

    // ---- workspace bump allocator ----
    char* wsb = (char*)d_ws; size_t off = 0;
    auto aw = [&](size_t bytes) -> void* {
        void* p = wsb + off; off = (off + bytes + 255) & ~(size_t)255; return p;
    };
    float* X    = (float*)aw((size_t)M4 * DM * 4);
    float* Xn   = (float*)aw((size_t)M4 * DM * 4);
    u16*   Xnb  = (u16*)  aw((size_t)M4 * DM * 2);
    float* QKV  = (float*)aw((size_t)M4 * D3 * 4);
    u16*   QKVb = (u16*)  aw((size_t)M4 * D3 * 2);
    float* SC   = (float*)aw((size_t)ZBH * SS * SS * 4);
    u16*   Pb   = (u16*)  aw((size_t)ZBH * SS * SS * 2);
    u16*   AOb  = (u16*)  aw((size_t)M4 * DM * 2);
    u16*   Hb   = (u16*)  aw((size_t)M4 * FFD * 2);
    u16*   MEMb = (u16*)  aw((size_t)M4 * DM * 2);
    u16*   Qcb  = (u16*)  aw((size_t)M4 * DM * 2);
    u16*   Kcb  = (u16*)  aw((size_t)M4 * DM * 2);
    float* Vc   = (float*)aw((size_t)M4 * DM * 4);

    const long sS3 = (long)SS * D3;     // per-batch stride inside QKV
    const long sSD = (long)SS * DM;
    const long sHS = (long)HH * SS * SS;
    const long sSs = (long)SS * SS;

    // ======================= ENCODER =======================
    embed_k<<<M4, 256, 0, stream>>>(src, embS, X);
    for (int i = 0; i < NLAY; ++i) {
        const float* qkvW = F(E_QKV_W) + (size_t)i * D3 * DM;
        const float* qkvB = F(E_QKV_B) + (size_t)i * D3;
        const float* prW  = F(E_PROJ_W) + (size_t)i * DM * DM;
        const float* prB  = F(E_PROJ_B) + (size_t)i * DM;
        const float* w1W  = F(E_W1) + (size_t)i * FFD * DM;
        const float* w1B  = F(E_B1) + (size_t)i * FFD;
        const float* w2W  = F(E_W2) + (size_t)i * DM * FFD;
        const float* w2B  = F(E_B2) + (size_t)i * DM;

        ln_k<<<M4, 256, 0, stream>>>(X, F(E_LN0A)+i*DM, F(E_LN0B)+i*DM, Xn, Xnb);
        run_gemm_nt(stream, false, GP(Xnb, DM,0,0, qkvW, DM,0,0, qkvB, nullptr,0,
                                      QKV, QKVb, D3,0,0, M4, D3, DM, 1,0,1.0f), 1);
        run_gemm_nt(stream, true,  GP(QKVb, D3, sS3, DH, QKVb + DM, D3, sS3, DH,
                                      nullptr, nullptr,0, SC, nullptr, SS, sHS, sSs,
                                      SS, SS, DH, HH, 0, 0.125f), ZBH);
        softmax_k<<<dim3(SS, ZBH), 256, 0, stream>>>(SC, Pb, 0);
        run_gemm_nn(stream, GP(Pb, SS, sHS, sSs, QKV + 2*DM, D3, sS3, DH,
                               nullptr, nullptr,0, nullptr, AOb, DM, sSD, DH,
                               SS, DH, SS, HH, 0, 1.0f), ZBH);
        run_gemm_nt(stream, false, GP(AOb, DM,0,0, prW, DM,0,0, prB, Xn, DM,
                                      X, nullptr, DM,0,0, M4, DM, DM, 1,0,1.0f), 1);
        ln_k<<<M4, 256, 0, stream>>>(X, F(E_LN1A)+i*DM, F(E_LN1B)+i*DM, Xn, Xnb);
        run_gemm_nt(stream, false, GP(Xnb, DM,0,0, w1W, DM,0,0, w1B, nullptr,0,
                                      nullptr, Hb, FFD,0,0, M4, FFD, DM, 1,1,1.0f), 1);
        run_gemm_nt(stream, false, GP(Hb, FFD,0,0, w2W, FFD,0,0, w2B, Xn, DM,
                                      X, nullptr, DM,0,0, M4, DM, FFD, 1,0,1.0f), 1);
        ln_k<<<M4, 256, 0, stream>>>(X, F(E_LN2A)+i*DM, F(E_LN2B)+i*DM,
                                     X, (i == NLAY-1) ? MEMb : Xnb);
    }

    // ======================= DECODER =======================
    embed_k<<<M4, 256, 0, stream>>>(tgt, embT, X);
    for (int i = 0; i < NLAY; ++i) {
        const float* qkvW = F(D_QKV_W) + (size_t)i * D3 * DM;
        const float* qkvB = F(D_QKV_B) + (size_t)i * D3;
        const float* spW  = F(D_SPROJ_W) + (size_t)i * DM * DM;
        const float* spB  = F(D_SPROJ_B) + (size_t)i * DM;
        const float* wqW  = F(D_WQ_W) + (size_t)i * DM * DM;
        const float* wqB  = F(D_WQ_B) + (size_t)i * DM;
        const float* wkW  = F(D_WK_W) + (size_t)i * DM * DM;
        const float* wkB  = F(D_WK_B) + (size_t)i * DM;
        const float* wvW  = F(D_WV_W) + (size_t)i * DM * DM;
        const float* wvB  = F(D_WV_B) + (size_t)i * DM;
        const float* cpW  = F(D_CPROJ_W) + (size_t)i * DM * DM;
        const float* cpB  = F(D_CPROJ_B) + (size_t)i * DM;
        const float* w1W  = F(D_W1) + (size_t)i * FFD * DM;
        const float* w1B  = F(D_B1) + (size_t)i * FFD;
        const float* w2W  = F(D_W2) + (size_t)i * DM * FFD;
        const float* w2B  = F(D_B2) + (size_t)i * DM;

        // -- masked self attention --
        ln_k<<<M4, 256, 0, stream>>>(X, F(D_LN0A)+i*DM, F(D_LN0B)+i*DM, Xn, Xnb);
        run_gemm_nt(stream, false, GP(Xnb, DM,0,0, qkvW, DM,0,0, qkvB, nullptr,0,
                                      QKV, QKVb, D3,0,0, M4, D3, DM, 1,0,1.0f), 1);
        run_gemm_nt(stream, true,  GP(QKVb, D3, sS3, DH, QKVb + DM, D3, sS3, DH,
                                      nullptr, nullptr,0, SC, nullptr, SS, sHS, sSs,
                                      SS, SS, DH, HH, 0, 0.125f), ZBH);
        softmax_k<<<dim3(SS, ZBH), 256, 0, stream>>>(SC, Pb, 1);
        run_gemm_nn(stream, GP(Pb, SS, sHS, sSs, QKV + 2*DM, D3, sS3, DH,
                               nullptr, nullptr,0, nullptr, AOb, DM, sSD, DH,
                               SS, DH, SS, HH, 0, 1.0f), ZBH);
        run_gemm_nt(stream, false, GP(AOb, DM,0,0, spW, DM,0,0, spB, Xn, DM,
                                      X, nullptr, DM,0,0, M4, DM, DM, 1,0,1.0f), 1);
        // -- cross attention --
        ln_k<<<M4, 256, 0, stream>>>(X, F(D_LN1A)+i*DM, F(D_LN1B)+i*DM, Xn, Xnb);
        run_gemm_nt(stream, false, GP(Xnb, DM,0,0, wqW, DM,0,0, wqB, nullptr,0,
                                      nullptr, Qcb, DM,0,0, M4, DM, DM, 1,0,1.0f), 1);
        run_gemm_nt(stream, false, GP(MEMb, DM,0,0, wkW, DM,0,0, wkB, nullptr,0,
                                      nullptr, Kcb, DM,0,0, M4, DM, DM, 1,0,1.0f), 1);
        run_gemm_nt(stream, false, GP(MEMb, DM,0,0, wvW, DM,0,0, wvB, nullptr,0,
                                      Vc, nullptr, DM,0,0, M4, DM, DM, 1,0,1.0f), 1);
        run_gemm_nt(stream, true,  GP(Qcb, DM, sSD, DH, Kcb, DM, sSD, DH,
                                      nullptr, nullptr,0, SC, nullptr, SS, sHS, sSs,
                                      SS, SS, DH, HH, 0, 0.125f), ZBH);
        softmax_k<<<dim3(SS, ZBH), 256, 0, stream>>>(SC, Pb, 0);
        run_gemm_nn(stream, GP(Pb, SS, sHS, sSs, Vc, DM, sSD, DH,
                               nullptr, nullptr,0, nullptr, AOb, DM, sSD, DH,
                               SS, DH, SS, HH, 0, 1.0f), ZBH);
        run_gemm_nt(stream, false, GP(AOb, DM,0,0, cpW, DM,0,0, cpB, Xn, DM,
                                      X, nullptr, DM,0,0, M4, DM, DM, 1,0,1.0f), 1);
        // -- FFN --
        ln_k<<<M4, 256, 0, stream>>>(X, F(D_LN2A)+i*DM, F(D_LN2B)+i*DM, Xn, Xnb);
        run_gemm_nt(stream, false, GP(Xnb, DM,0,0, w1W, DM,0,0, w1B, nullptr,0,
                                      nullptr, Hb, FFD,0,0, M4, FFD, DM, 1,1,1.0f), 1);
        run_gemm_nt(stream, false, GP(Hb, FFD,0,0, w2W, FFD,0,0, w2B, Xn, DM,
                                      X, nullptr, DM,0,0, M4, DM, FFD, 1,0,1.0f), 1);
        ln_k<<<M4, 256, 0, stream>>>(X, F(D_LN3A)+i*DM, F(D_LN3B)+i*DM,
                                     (i == NLAY-1) ? (float*)d_out : X, Xnb);
    }
}
